// LSTMCell_4123168604852
// MI455X (gfx1250) — matrix operand order repaired and checked by reading, not compiled, not run on hardware
//
#include <hip/hip_runtime.h>

// ---------------------------------------------------------------------------
// LSTM cell, fused bf16-WMMA implementation for gfx1250 (MI455X).
//   gate_inputs = [x | hidden]            : [2048 x 2048]
//   pre_g = gate_inputs @ W_g + b_g       : 4 GEMMs M=2048,N=1024,K=2048
//   c' = sigm(f)*c + sigm(i)*tanh(c_hat) ; h' = sigm(o)*tanh(c')
//
// GEMM kernel: BM=128 x BN=32 tile, BK=32, 8 waves (4 M-waves x 2 N-waves),
// each wave: 2 A fragments x 4 gates = 8 WMMAs / K-step. Double-buffered LDS
// (power-of-2 stride, XOR toggle), one barrier per K-step, unroll 1.
// Staging uses gfx1250 async load-to-LDS (ASYNCcnt-tracked, no staging
// VGPRs) when the builtin is available, else global->VGPR->ds_store.
// ---------------------------------------------------------------------------

#define B_DIM   2048
#define IN_DIM  1024
#define H_DIM   1024
#define K_DIM   2048          // IN + H

#define BM 128
#define BN 32
#define BK 32
#define LDA 40                // padded LDS row stride (elems) -> conflict-free b128
#define LDB 40
#define KSTEPS (K_DIM / BK)   // 64
#define ABUF 8192             // elems per LDS buffer (16 KB), power of two
#define BBUF 8192

#if defined(__AMDGCN__) && __has_builtin(__builtin_amdgcn_global_load_async_to_lds_b128)
#define HAVE_ASYNC 1
#endif

#ifdef HAVE_ASYNC
typedef __attribute__((__vector_size__(4 * sizeof(int)))) int i32x4_t;
#define AS_GLOBAL(p) ((__attribute__((address_space(1))) i32x4_t*)(p))
#define AS_LDS(p)    ((__attribute__((address_space(3))) i32x4_t*)(p))
#if __has_builtin(__builtin_amdgcn_s_wait_asynccnt)
#define ASYNC_WAIT() __builtin_amdgcn_s_wait_asynccnt(0)
#else
#define ASYNC_WAIT() asm volatile("s_wait_asynccnt 0x0" ::: "memory")
#endif
#endif

typedef __attribute__((ext_vector_type(16))) __bf16 bf16x16;
typedef __attribute__((ext_vector_type(8)))  float  f32x8;

union FragU { uint4 u[2]; bf16x16 v; };

__device__ __forceinline__ unsigned short f2bf(float f) {
    unsigned int u = __float_as_uint(f);
    u += 0x7FFFu + ((u >> 16) & 1u);      // round-to-nearest-even
    return (unsigned short)(u >> 16);
}
__device__ __forceinline__ float sigm(float x) {
    return 1.0f / (1.0f + __expf(-x));
}
__device__ __forceinline__ float tanh_fast(float x) {
    return 1.0f - 2.0f / (1.0f + __expf(2.0f * x));
}

// --------------------------------------------------------------------------
// Kernel 1: pack concatenated activations [x|hidden] -> bf16 row-major [B,K]
// --------------------------------------------------------------------------
__global__ __launch_bounds__(256) void pack_a(const float* __restrict__ x,
                                              const float* __restrict__ hid,
                                              unsigned short* __restrict__ A) {
    int e    = blockIdx.x * 256 + threadIdx.x;   // each thread: 8 elements
    int base = e * 8;
    int b    = base >> 11;                       // /2048
    int k    = base & 2047;
    const float* src = (k < IN_DIM) ? (x + (size_t)b * IN_DIM + k)
                                    : (hid + (size_t)b * H_DIM + (k - IN_DIM));
    const float4* p = (const float4*)src;
    float4 f0 = p[0];
    float4 f1 = p[1];
    uint4 o;
    o.x = (unsigned)f2bf(f0.x) | ((unsigned)f2bf(f0.y) << 16);
    o.y = (unsigned)f2bf(f0.z) | ((unsigned)f2bf(f0.w) << 16);
    o.z = (unsigned)f2bf(f1.x) | ((unsigned)f2bf(f1.y) << 16);
    o.w = (unsigned)f2bf(f1.z) | ((unsigned)f2bf(f1.w) << 16);
    *(uint4*)(A + base) = o;
}

// --------------------------------------------------------------------------
// Kernel 2: transpose-convert weights W[k][n] f32 -> Wt[g][n][k] bf16
// --------------------------------------------------------------------------
__global__ __launch_bounds__(256) void pack_w(const float* __restrict__ Wf,
                                              const float* __restrict__ Wi,
                                              const float* __restrict__ Wc,
                                              const float* __restrict__ Wo,
                                              unsigned short* __restrict__ Wt) {
    __shared__ float tile[32][33];
    int g = blockIdx.z;
    const float* W = (g == 0) ? Wf : (g == 1) ? Wi : (g == 2) ? Wc : Wo;
    int kt = blockIdx.x;     // K tile (64)
    int nt = blockIdx.y;     // N tile (32)
    int tc = threadIdx.x & 31;
    int tr = threadIdx.x >> 5;        // 0..7
#pragma unroll
    for (int i = 0; i < 4; ++i) {
        int kr = tr + i * 8;
        tile[kr][tc] = W[(size_t)(kt * 32 + kr) * H_DIM + nt * 32 + tc];
    }
    __syncthreads();
    unsigned short* dst = Wt + (size_t)g * H_DIM * K_DIM;
#pragma unroll
    for (int i = 0; i < 4; ++i) {
        int nr = tr + i * 8;
        dst[(size_t)(nt * 32 + nr) * K_DIM + kt * 32 + tc] = f2bf(tile[tc][nr]);
    }
}

// One K-step of compute from LDS buffers Ab/Bb (8 WMMAs per wave).
#define COMPUTE_TILE(Ab, Bb)                                                   \
    do {                                                                       \
        FragU a0, a1;                                                          \
        a0.u[0] = *(const uint4*)&(Ab)[afo0];                                  \
        a0.u[1] = *(const uint4*)&(Ab)[afo0 + 8];                              \
        a1.u[0] = *(const uint4*)&(Ab)[afo1];                                  \
        a1.u[1] = *(const uint4*)&(Ab)[afo1 + 8];                              \
        FragU bfr;                                                             \
        bfr.u[0] = *(const uint4*)&(Bb)[fbo];                                  \
        bfr.u[1] = *(const uint4*)&(Bb)[fbo + 8];                              \
        accF0 = __builtin_amdgcn_wmma_f32_16x16x32_bf16(                       \
            false, a0.v, false, bfr.v, (short)0, accF0, false, false);         \
        accF1 = __builtin_amdgcn_wmma_f32_16x16x32_bf16(                       \
            false, a1.v, false, bfr.v, (short)0, accF1, false, false);         \
        bfr.u[0] = *(const uint4*)&(Bb)[1 * BK * LDB + fbo];                   \
        bfr.u[1] = *(const uint4*)&(Bb)[1 * BK * LDB + fbo + 8];               \
        accI0 = __builtin_amdgcn_wmma_f32_16x16x32_bf16(                       \
            false, a0.v, false, bfr.v, (short)0, accI0, false, false);         \
        accI1 = __builtin_amdgcn_wmma_f32_16x16x32_bf16(                       \
            false, a1.v, false, bfr.v, (short)0, accI1, false, false);         \
        bfr.u[0] = *(const uint4*)&(Bb)[2 * BK * LDB + fbo];                   \
        bfr.u[1] = *(const uint4*)&(Bb)[2 * BK * LDB + fbo + 8];               \
        accC0 = __builtin_amdgcn_wmma_f32_16x16x32_bf16(                       \
            false, a0.v, false, bfr.v, (short)0, accC0, false, false);         \
        accC1 = __builtin_amdgcn_wmma_f32_16x16x32_bf16(                       \
            false, a1.v, false, bfr.v, (short)0, accC1, false, false);         \
        bfr.u[0] = *(const uint4*)&(Bb)[3 * BK * LDB + fbo];                   \
        bfr.u[1] = *(const uint4*)&(Bb)[3 * BK * LDB + fbo + 8];               \
        accO0 = __builtin_amdgcn_wmma_f32_16x16x32_bf16(                       \
            false, a0.v, false, bfr.v, (short)0, accO0, false, false);         \
        accO1 = __builtin_amdgcn_wmma_f32_16x16x32_bf16(                       \
            false, a1.v, false, bfr.v, (short)0, accO1, false, false);         \
    } while (0)

#ifdef HAVE_ASYNC
// Async (ASYNCcnt-tracked) staging of one K-tile directly into LDS.
#define STAGE_TILE(koff, nb)                                                   \
    do {                                                                       \
        __builtin_amdgcn_global_load_async_to_lds_b128(                        \
            AS_GLOBAL(agp0 + (koff)), AS_LDS(AsF + (nb) + aso0), 0, 0);        \
        __builtin_amdgcn_global_load_async_to_lds_b128(                        \
            AS_GLOBAL(agp1 + (koff)), AS_LDS(AsF + (nb) + aso1), 0, 0);        \
        __builtin_amdgcn_global_load_async_to_lds_b128(                        \
            AS_GLOBAL(bgp0 + (koff)), AS_LDS(BsF + (nb) + bso0), 0, 0);        \
        __builtin_amdgcn_global_load_async_to_lds_b128(                        \
            AS_GLOBAL(bgp1 + (koff)), AS_LDS(BsF + (nb) + bso1), 0, 0);        \
    } while (0)
#endif

// --------------------------------------------------------------------------
// Kernel 3: fused 4-gate GEMM + LSTM epilogue (double-buffered LDS).
// --------------------------------------------------------------------------
__global__ __launch_bounds__(256) void lstm_gemm(
        const unsigned short* __restrict__ A,    // [B, K] bf16
        const unsigned short* __restrict__ Wt,   // [4][H, K] bf16
        const float* __restrict__ bf_, const float* __restrict__ bi_,
        const float* __restrict__ bc_, const float* __restrict__ bo_,
        const float* __restrict__ cell, float* __restrict__ out) {
    __shared__ __align__(16) unsigned short AsF[2 * ABUF];   // 32 KB (10 KB used/buf)
    __shared__ __align__(16) unsigned short BsF[2 * BBUF];   // 32 KB (10 KB used/buf)

    const int t  = threadIdx.x;
    const int w  = t >> 5;
    const int L  = t & 31;
    const int wm = w & 3;            // 0..3 -> 32-row M sub-tile
    const int wn = w >> 2;           // 0..1 -> 16-col N sub-tile
    const int m0 = blockIdx.x * BM;
    const int n0 = blockIdx.y * BN;

    f32x8 accF0 = {}, accI0 = {}, accC0 = {}, accO0 = {};
    f32x8 accF1 = {}, accI1 = {}, accC1 = {}, accO1 = {};

    // ---- staging: A tile 128x32 (512 b128 chunks), B tiles 4x32x32 (512) ----
    const int ar0 = t >> 2,          akg = t & 3;
    const int ar1 = (t + 256) >> 2;
    const unsigned short* agp0 = A + (size_t)(m0 + ar0) * K_DIM + akg * 8;
    const unsigned short* agp1 = A + (size_t)(m0 + ar1) * K_DIM + akg * 8;
    const int aso0 = ar0 * LDA + akg * 8;
    const int aso1 = ar1 * LDA + akg * 8;

    const int g0 = t >> 7,        g1 = (t + 256) >> 7;
    const int n0c = (t & 127) >> 2;
    const unsigned short* bgp0 =
        Wt + (size_t)g0 * H_DIM * K_DIM + (size_t)(n0 + n0c) * K_DIM + akg * 8;
    const unsigned short* bgp1 =
        Wt + (size_t)g1 * H_DIM * K_DIM + (size_t)(n0 + n0c) * K_DIM + akg * 8;
    const int bso0 = g0 * BK * LDB + n0c * LDB + akg * 8;
    const int bso1 = g1 * BK * LDB + n0c * LDB + akg * 8;

    // ---- fragment offsets (A and B both use contiguous k = 16*half + s) ----
    const int half = L >> 4;
    const int lane16 = L & 15;
    const int afo0 = (wm * 32 + lane16) * LDA + half * 16;       // A frag rows 0..15
    const int afo1 = (wm * 32 + 16 + lane16) * LDA + half * 16;  // A frag rows 16..31
    const int fbo  = (wn * 16 + lane16) * LDB + half * 16;       // B frag (+g*BK*LDB)

    // ---- prologue: stage K-tile 0 into buffer 0 ----
#ifdef HAVE_ASYNC
    STAGE_TILE(0, 0);
    ASYNC_WAIT();
#else
    {
        uint4 av0 = *(const uint4*)agp0;
        uint4 av1 = *(const uint4*)agp1;
        uint4 bv0 = *(const uint4*)bgp0;
        uint4 bv1 = *(const uint4*)bgp1;
        *(uint4*)&AsF[aso0] = av0;
        *(uint4*)&AsF[aso1] = av1;
        *(uint4*)&BsF[bso0] = bv0;
        *(uint4*)&BsF[bso1] = bv1;
    }
#endif
    __syncthreads();

    // ---- steady state: stage next tile, compute current, barrier ----
#pragma unroll 1
    for (int it = 0; it < KSTEPS - 1; ++it) {
        const int cb = (it & 1) ? ABUF : 0;   // current buffer offset (elems)
        const int nb = cb ^ ABUF;             // next buffer offset
        const unsigned short* Ab = AsF + cb;
        const unsigned short* Bb = BsF + cb;
        const int k1 = (it + 1) * BK;

#ifdef HAVE_ASYNC
        // async L2->LDS for tile it+1; runs behind the WMMA stream
        STAGE_TILE(k1, nb);
        COMPUTE_TILE(Ab, Bb);
        ASYNC_WAIT();
#else
        uint4 av0 = *(const uint4*)(agp0 + k1);
        uint4 av1 = *(const uint4*)(agp1 + k1);
        uint4 bv0 = *(const uint4*)(bgp0 + k1);
        uint4 bv1 = *(const uint4*)(bgp1 + k1);

        COMPUTE_TILE(Ab, Bb);

        // target buffer's readers finished at the barrier ending iteration it-1
        *(uint4*)&AsF[nb + aso0] = av0;
        *(uint4*)&AsF[nb + aso1] = av1;
        *(uint4*)&BsF[nb + bso0] = bv0;
        *(uint4*)&BsF[nb + bso1] = bv1;
#endif
        __syncthreads();
    }

    // ---- peeled last K-step: tile KSTEPS-1 lives in buffer 1 ----
    {
        const unsigned short* Ab = AsF + ABUF;
        const unsigned short* Bb = BsF + BBUF;
        COMPUTE_TILE(Ab, Bb);
    }

    // ---- fused LSTM epilogue ----
    // C/D layout: lane = col n (0..15), VGPR r = row half*8 + r
    const int h = n0 + wn * 16 + lane16;
    const float vbf = bf_[h], vbi = bi_[h], vbc = bc_[h], vbo = bo_[h];
    const int rowBase = m0 + wm * 32 + half * 8;
#pragma unroll
    for (int r = 0; r < 8; ++r) {
        const size_t idx = (size_t)(rowBase + r) * H_DIM + h;
        float f  = sigm(accF0[r] + vbf);
        float i  = sigm(accI0[r] + vbi);
        float ch = tanh_fast(accC0[r] + vbc);
        float o  = sigm(accO0[r] + vbo);
        float cn = f * cell[idx] + i * ch;
        out[idx] = cn;                                         // c_next
        out[idx + (size_t)B_DIM * H_DIM] = o * tanh_fast(cn);  // h_next
    }
#pragma unroll
    for (int r = 0; r < 8; ++r) {
        const size_t idx = (size_t)(rowBase + 16 + r) * H_DIM + h;
        float f  = sigm(accF1[r] + vbf);
        float i  = sigm(accI1[r] + vbi);
        float ch = tanh_fast(accC1[r] + vbc);
        float o  = sigm(accO1[r] + vbo);
        float cn = f * cell[idx] + i * ch;
        out[idx] = cn;                                         // c_next
        out[idx + (size_t)B_DIM * H_DIM] = o * tanh_fast(cn);  // h_next
    }
}

// --------------------------------------------------------------------------
extern "C" void kernel_launch(void* const* d_in, const int* in_sizes, int n_in,
                              void* d_out, int out_size, void* d_ws, size_t ws_size,
                              hipStream_t stream) {
    const float* x    = (const float*)d_in[0];
    const float* cell = (const float*)d_in[1];
    const float* hid  = (const float*)d_in[2];
    const float* Wf   = (const float*)d_in[3];
    const float* bf_  = (const float*)d_in[4];
    const float* Wi   = (const float*)d_in[5];
    const float* bi_  = (const float*)d_in[6];
    const float* Wc   = (const float*)d_in[7];
    const float* bc_  = (const float*)d_in[8];
    const float* Wo   = (const float*)d_in[9];
    const float* bo_  = (const float*)d_in[10];

    unsigned short* Abf = (unsigned short*)d_ws;               // 8 MB
    unsigned short* Wt  = Abf + (size_t)B_DIM * K_DIM;         // 16 MB

    // 1) activations -> bf16 [B,K]
    pack_a<<<(B_DIM * K_DIM / 8) / 256, 256, 0, stream>>>(x, hid, Abf);
    // 2) weights -> bf16 transposed [4][H,K]
    pack_w<<<dim3(K_DIM / 32, H_DIM / 32, 4), 256, 0, stream>>>(Wf, Wi, Wc, Wo, Wt);
    // 3) fused 4-gate WMMA GEMM + LSTM epilogue
    lstm_gemm<<<dim3(B_DIM / BM, H_DIM / BN), 256, 0, stream>>>(
        Abf, Wt, bf_, bi_, bc_, bo_, cell, (float*)d_out);
}
